// Encoder_67095979099046
// MI455X (gfx1250) — matrix operand order repaired and checked by reading, not compiled, not run on hardware
//
#include <hip/hip_runtime.h>
#include <hip/hip_bf16.h>

// ---------------------------------------------------------------------------
// Problem constants (match reference)
// ---------------------------------------------------------------------------
#define N0v 300000
#define N1v 60000
#define N2v 12000
#define E0v 1500000
#define E1v 300000
#define HIDv 128
#define Rv 3
#define Hv 4
#define OUTv 153
#define BN_EPSv 1e-5f

typedef __attribute__((ext_vector_type(16))) __bf16 v16bf;
typedef __attribute__((ext_vector_type(8)))  __bf16 v8bf;
typedef __attribute__((ext_vector_type(8)))  float  v8f;

// monotone float<->int encoding for atomicMax on signed ints
__device__ __forceinline__ int enc_f(float f) {
    int i = __builtin_bit_cast(int, f);
    return (i >= 0) ? i : (i ^ 0x7FFFFFFF);
}
__device__ __forceinline__ float dec_f(int i) {
    int j = (i >= 0) ? i : (i ^ 0x7FFFFFFF);
    return __builtin_bit_cast(float, j);
}

__device__ __forceinline__ v16bf pack16(float4 c0, float4 c1, float4 c2, float4 c3) {
    v16bf v;
    v[0]  = (__bf16)c0.x; v[1]  = (__bf16)c0.y; v[2]  = (__bf16)c0.z; v[3]  = (__bf16)c0.w;
    v[4]  = (__bf16)c1.x; v[5]  = (__bf16)c1.y; v[6]  = (__bf16)c1.z; v[7]  = (__bf16)c1.w;
    v[8]  = (__bf16)c2.x; v[9]  = (__bf16)c2.y; v[10] = (__bf16)c2.z; v[11] = (__bf16)c2.w;
    v[12] = (__bf16)c3.x; v[13] = (__bf16)c3.y; v[14] = (__bf16)c3.z; v[15] = (__bf16)c3.w;
    return v;
}

// ---------------------------------------------------------------------------
// GEMM: C[M x Ncols] = A[M x 128] @ B[128 x Ncols] (+bias +sum3(bias3))
// bf16 WMMA, fp32 accumulate.
//  - B staged once per block in LDS (transposed, bf16, zero-padded columns).
//  - Combined bias vector staged in LDS -> branch-free epilogue.
//  - Each wave owns 32 rows (two 16-row M tiles); per N tile we preload all
//    4 B fragments (8x ds_load_b128, one dscnt wait) then run the 8-WMMA
//    chain on two independent accumulators.
// Fragment layouts per CDNA5 ISA 7.12.2:
//  A 16x32: lane&15 = M; half=lane>>4 -> K chunks {b..b+7},{b+16..b+23}.
//  B 32x16: lane&15 = N; half -> same K chunks as A.
//  C tile:  elem j -> row (half*8+j), col lane&15.
// ---------------------------------------------------------------------------
#define BT_STRIDE 136      // 128 + 8 halfs pad: 272B row -> conflict-free b128
#define BT_MAXN   160      // max padded columns (ceil(153/16)*16)

__global__ __launch_bounds__(256) void gemm128_bf16_wmma(
    const float* __restrict__ A, const float* __restrict__ B,
    const float* __restrict__ bias, const float* __restrict__ bias3,
    float* __restrict__ C, int M, int Ncols, int ldb, int ldo)
{
    __shared__ __bf16 Bt[BT_MAXN * BT_STRIDE];
    __shared__ float  biasLds[BT_MAXN];

    int ntiles = (Ncols + 15) >> 4;
    int npad   = ntiles << 4;

    // Stage combined bias (branch-free epilogue later).
    for (int n = threadIdx.x; n < npad; n += 256) {
        float bb = 0.0f;
        if (n < Ncols) {
            if (bias)  bb += bias[n];
            if (bias3) bb += bias3[n] + bias3[128 + n] + bias3[256 + n];
        }
        biasLds[n] = bb;
    }

    // Cooperative stage: Bt[n][k] = bf16(B[k*ldb + n]); zero-fill n >= Ncols.
    // Consecutive threads touch consecutive n -> coalesced global reads.
    for (int idx = threadIdx.x; idx < (npad << 7); idx += 256) {
        int n = idx % npad;
        int k = idx / npad;
        float v = (n < Ncols) ? B[(size_t)k * ldb + n] : 0.0f;
        Bt[n * BT_STRIDE + k] = (__bf16)v;
    }
    __syncthreads();

    int wave = blockIdx.x * 8 + (threadIdx.x >> 5);
    int m0 = wave << 5;                       // 32 rows per wave
    if (m0 >= M) return;                      // M always a multiple of 32 here
    int lane = threadIdx.x & 31;
    int mr   = lane & 15;
    int half = lane >> 4;

    // Load + convert K=128 A strips for the two M tiles (b128 loads).
    v16bf a0[4], a1[4];
    const float4* r0 = (const float4*)(A + (size_t)(m0 + mr) * 128);
    const float4* r1 = (const float4*)(A + (size_t)(m0 + 16 + mr) * 128);
    #pragma unroll
    for (int ks = 0; ks < 4; ++ks) {
        int f4 = ks * 8 + half * 2;
        a0[ks] = pack16(r0[f4], r0[f4 + 1], r0[f4 + 4], r0[f4 + 5]);
        a1[ks] = pack16(r1[f4], r1[f4 + 1], r1[f4 + 4], r1[f4 + 5]);
    }

    for (int ct = 0; ct < ntiles; ++ct) {
        int n = (ct << 4) + mr;

        // Preload all 4 B fragments for this N tile (8x ds_load_b128).
        v16bf bfr[4];
        #pragma unroll
        for (int ks = 0; ks < 4; ++ks) {
            const v8bf* bp = (const v8bf*)&Bt[n * BT_STRIDE + ks * 32 + half * 8];
            union { v16bf v; v8bf h[2]; } u;
            u.h[0] = bp[0];
            u.h[1] = bp[2];
            bfr[ks] = u.v;
        }

        v8f acc0 = {}, acc1 = {};
        #pragma unroll
        for (int ks = 0; ks < 4; ++ks) {
            acc0 = __builtin_amdgcn_wmma_f32_16x16x32_bf16(
                false, a0[ks], false, bfr[ks], (short)0, acc0, false, false);
            acc1 = __builtin_amdgcn_wmma_f32_16x16x32_bf16(
                false, a1[ks], false, bfr[ks], (short)0, acc1, false, false);
        }

        if (n < Ncols) {
            float bb = biasLds[n];
            #pragma unroll
            for (int j = 0; j < 8; ++j) {
                int rowj = m0 + half * 8 + j;
                C[(size_t)rowj * ldo + n]        = acc0[j] + bb;
                C[(size_t)(rowj + 16) * ldo + n] = acc1[j] + bb;
            }
        }
    }
}

// a[n,h] = sum_c hs[n, h*32+c] * att[h*32+c]
__global__ __launch_bounds__(256) void rowdot_att(
    const float* __restrict__ hs, const float* __restrict__ att,
    float* __restrict__ out, int n)
{
    int idx = blockIdx.x * blockDim.x + threadIdx.x;
    if (idx >= n * Hv) return;
    int node = idx >> 2, h = idx & 3;
    const float4* p = (const float4*)(hs + (size_t)node * HIDv + h * 32);
    const float4* a = (const float4*)(att + h * 32);
    float s = 0.0f;
    #pragma unroll
    for (int c = 0; c < 8; ++c) {
        float4 pv = p[c], av = a[c];
        s += pv.x * av.x + pv.y * av.y + pv.z * av.z + pv.w * av.w;
    }
    out[idx] = s;
}

__global__ __launch_bounds__(256) void fill_u32(unsigned* __restrict__ p, unsigned v, int n)
{
    int i = blockIdx.x * blockDim.x + threadIdx.x;
    if (i < n) p[i] = v;
}

// Edge pass 1: segment max of leaky_relu(a_s[col]+a_t[row]) over (row, head)
__global__ __launch_bounds__(256) void edge_max(
    const int* __restrict__ row, const int* __restrict__ col,
    const int* __restrict__ et, int E, int r,
    const float* __restrict__ as_, const float* __restrict__ at_,
    int* __restrict__ m)
{
    int e = blockIdx.x * blockDim.x + threadIdx.x;
    if (e >= E || et[e] != r) return;
    int rr = row[e], cc = col[e];
    #pragma unroll
    for (int h = 0; h < Hv; ++h) {
        float v = as_[cc * Hv + h] + at_[rr * Hv + h];
        v = (v > 0.0f) ? v : 0.2f * v;          // leaky_relu 0.2
        atomicMax(&m[rr * Hv + h], enc_f(v));
    }
}

// Edge pass 2: ez = exp(e - max), segment sum of ez
__global__ __launch_bounds__(256) void edge_expsum(
    const int* __restrict__ row, const int* __restrict__ col,
    const int* __restrict__ et, int E, int r,
    const float* __restrict__ as_, const float* __restrict__ at_,
    const int* __restrict__ m, float* __restrict__ den,
    float* __restrict__ ez)
{
    int e = blockIdx.x * blockDim.x + threadIdx.x;
    if (e >= E || et[e] != r) return;
    int rr = row[e], cc = col[e];
    #pragma unroll
    for (int h = 0; h < Hv; ++h) {
        float v = as_[cc * Hv + h] + at_[rr * Hv + h];
        v = (v > 0.0f) ? v : 0.2f * v;
        float z = __expf(v - dec_f(m[rr * Hv + h]));
        ez[(size_t)e * Hv + h] = z;
        atomicAdd(&den[rr * Hv + h], z);
    }
}

// Edge pass 3: out[row,:] += (ez/den) * hs[col,:]   (one wave per edge)
__global__ __launch_bounds__(256) void edge_scatter(
    const int* __restrict__ row, const int* __restrict__ col,
    const int* __restrict__ et, int E, int r,
    const float* __restrict__ ez, const float* __restrict__ den,
    const float* __restrict__ hs, float* __restrict__ out)
{
    int e = blockIdx.x * 8 + (threadIdx.x >> 5);
    if (e >= E || et[e] != r) return;
    int lane = threadIdx.x & 31;
    int rr = row[e], cc = col[e];
    const float* src = hs + (size_t)cc * HIDv;
    float* dst = out + (size_t)rr * HIDv;
    #pragma unroll
    for (int h = 0; h < Hv; ++h) {
        float w = ez[(size_t)e * Hv + h] / den[rr * Hv + h];
        int chan = h * 32 + lane;
        atomicAdd(&dst[chan], w * src[chan]);
    }
}

// BN pass 1: per-channel partial sums / sums-of-squares
__global__ __launch_bounds__(256) void bn_stats(
    const float* __restrict__ h, float* __restrict__ sums,
    float* __restrict__ sumsq, int n)
{
    __shared__ float s1[256], s2[256];
    int c = threadIdx.x & 127;
    int half = threadIdx.x >> 7;
    float a = 0.0f, b = 0.0f;
    for (int rowi = blockIdx.x * 2 + half; rowi < n; rowi += gridDim.x * 2) {
        float v = h[(size_t)rowi * HIDv + c];
        a += v; b += v * v;
    }
    s1[threadIdx.x] = a; s2[threadIdx.x] = b;
    __syncthreads();
    if (half == 0) {
        atomicAdd(&sums[c],  s1[c] + s1[c + 128]);
        atomicAdd(&sumsq[c], s2[c] + s2[c + 128]);
    }
}

// BN pass 2: normalize + gamma/beta + activation (0 = ELU, 1 = ReLU)
__global__ __launch_bounds__(256) void bn_apply(
    float* __restrict__ h, const float* __restrict__ sums,
    const float* __restrict__ sumsq, const float* __restrict__ g,
    const float* __restrict__ b, int n, int act)
{
    int idx = blockIdx.x * blockDim.x + threadIdx.x;
    if (idx >= n * HIDv) return;
    int c = idx & 127;
    float inv_n = 1.0f / (float)n;
    float mu  = sums[c] * inv_n;
    float var = sumsq[c] * inv_n - mu * mu;
    float y = (h[idx] - mu) * rsqrtf(var + BN_EPSv) * g[c] + b[c];
    h[idx] = act ? fmaxf(y, 0.0f) : ((y > 0.0f) ? y : (__expf(y) - 1.0f));
}

// ---------------------------------------------------------------------------
static inline int idiv(int a, int b) { return (a + b - 1) / b; }

extern "C" void kernel_launch(void* const* d_in, const int* in_sizes, int n_in,
                              void* d_out, int out_size, void* d_ws, size_t ws_size,
                              hipStream_t stream)
{
    const float* x      = (const float*)d_in[0];
    const int*   row0   = (const int*)d_in[1];
    const int*   col0   = (const int*)d_in[2];
    const int*   et0    = (const int*)d_in[3];
    const int*   row1   = (const int*)d_in[4];
    const int*   col1   = (const int*)d_in[5];
    const int*   et1    = (const int*)d_in[6];
    const float* Wsrc   = (const float*)d_in[9];
    const float* Wdst   = (const float*)d_in[10];
    const float* attS   = (const float*)d_in[11];
    const float* attD   = (const float*)d_in[12];
    const float* cbias  = (const float*)d_in[13];
    const float* skW    = (const float*)d_in[14];
    const float* skb    = (const float*)d_in[15];
    const float* bng    = (const float*)d_in[16];
    const float* bnb    = (const float*)d_in[17];
    const float* mW1    = (const float*)d_in[18];
    const float* mb1    = (const float*)d_in[19];
    const float* mbng   = (const float*)d_in[20];
    const float* mbnb   = (const float*)d_in[21];
    const float* mW2    = (const float*)d_in[22];
    const float* mb2    = (const float*)d_in[23];

    // -------- workspace carve-up --------
    char* wsp = (char*)d_ws;
    auto carve = [&](size_t bytes) -> void* {
        void* p = (void*)wsp;
        wsp += (bytes + 255) & ~(size_t)255;
        return p;
    };
    float* h1   = (float*)carve((size_t)N1v * HIDv * 4);   // layer0 output
    float* h2   = (float*)carve((size_t)N2v * HIDv * 4);   // layer1 output
    float* tbuf = (float*)carve((size_t)N2v * HIDv * 4);   // mlp hidden
    float* hs   = (float*)carve((size_t)N0v * HIDv * 4);   // per-relation src proj
    float* ht   = (float*)carve((size_t)N1v * HIDv * 4);   // per-relation tgt proj
    float* as_  = (float*)carve((size_t)N0v * Hv * 4);
    float* at_  = (float*)carve((size_t)N1v * Hv * 4);
    int*   mseg = (int*)  carve((size_t)N1v * Hv * 4);
    float* den  = (float*)carve((size_t)N1v * Hv * 4);
    float* ez   = (float*)carve((size_t)E0v * Hv * 4);
    float* sums = (float*)carve(HIDv * 4);
    float* sumq = (float*)carve(HIDv * 4);

    const int BLK = 256;

    for (int l = 0; l < 2; ++l) {
        const float* xin  = (l == 0) ? x : h1;
        float*       hout = (l == 0) ? h1 : h2;
        int n_src = (l == 0) ? N0v : N1v;
        int n_tgt = (l == 0) ? N1v : N2v;
        int E     = (l == 0) ? E0v : E1v;
        const int* row = (l == 0) ? row0 : row1;
        const int* col = (l == 0) ? col0 : col1;
        const int* et  = (l == 0) ? et0  : et1;

        // skip: hout = x_t @ skW[l] + skb[l] + sum_r conv_bias[l,r,:]
        gemm128_bf16_wmma<<<idiv(n_tgt, 256), BLK, 0, stream>>>(
            xin, skW + (size_t)l * HIDv * HIDv,
            skb + (size_t)l * HIDv, cbias + (size_t)l * Rv * HIDv,
            hout, n_tgt, HIDv, HIDv, HIDv);

        for (int r = 0; r < Rv; ++r) {
            const float* Ws = Wsrc + ((size_t)l * Rv + r) * HIDv * HIDv;
            const float* Wd = Wdst + ((size_t)l * Rv + r) * HIDv * HIDv;
            const float* aS = attS + ((size_t)l * Rv + r) * HIDv; // H*C = 128
            const float* aD = attD + ((size_t)l * Rv + r) * HIDv;

            gemm128_bf16_wmma<<<idiv(n_src, 256), BLK, 0, stream>>>(
                xin, Ws, nullptr, nullptr, hs, n_src, HIDv, HIDv, HIDv);
            rowdot_att<<<idiv(n_src * Hv, BLK), BLK, 0, stream>>>(hs, aS, as_, n_src);

            gemm128_bf16_wmma<<<idiv(n_tgt, 256), BLK, 0, stream>>>(
                xin, Wd, nullptr, nullptr, ht, n_tgt, HIDv, HIDv, HIDv);
            rowdot_att<<<idiv(n_tgt * Hv, BLK), BLK, 0, stream>>>(ht, aD, at_, n_tgt);

            fill_u32<<<idiv(n_tgt * Hv, BLK), BLK, 0, stream>>>(
                (unsigned*)mseg, 0x80000000u, n_tgt * Hv);
            fill_u32<<<idiv(n_tgt * Hv, BLK), BLK, 0, stream>>>(
                (unsigned*)den, 0u, n_tgt * Hv);

            edge_max<<<idiv(E, BLK), BLK, 0, stream>>>(
                row, col, et, E, r, as_, at_, mseg);
            edge_expsum<<<idiv(E, BLK), BLK, 0, stream>>>(
                row, col, et, E, r, as_, at_, mseg, den, ez);
            edge_scatter<<<idiv(E, 8), BLK, 0, stream>>>(
                row, col, et, E, r, ez, den, hs, hout);
        }

        // BatchNorm + ELU
        fill_u32<<<1, BLK, 0, stream>>>((unsigned*)sums, 0u, HIDv);
        fill_u32<<<1, BLK, 0, stream>>>((unsigned*)sumq, 0u, HIDv);
        bn_stats<<<256, BLK, 0, stream>>>(hout, sums, sumq, n_tgt);
        bn_apply<<<idiv(n_tgt * HIDv, BLK), BLK, 0, stream>>>(
            hout, sums, sumq, bng + (size_t)l * HIDv, bnb + (size_t)l * HIDv,
            n_tgt, /*act=*/0);
    }

    // MLP head
    gemm128_bf16_wmma<<<idiv(N2v, 256), BLK, 0, stream>>>(
        h2, mW1, mb1, nullptr, tbuf, N2v, HIDv, HIDv, HIDv);
    fill_u32<<<1, BLK, 0, stream>>>((unsigned*)sums, 0u, HIDv);
    fill_u32<<<1, BLK, 0, stream>>>((unsigned*)sumq, 0u, HIDv);
    bn_stats<<<256, BLK, 0, stream>>>(tbuf, sums, sumq, N2v);
    bn_apply<<<idiv(N2v * HIDv, BLK), BLK, 0, stream>>>(
        tbuf, sums, sumq, mbng, mbnb, N2v, /*act=*/1);

    gemm128_bf16_wmma<<<idiv(N2v, 256), BLK, 0, stream>>>(
        tbuf, mW2, mb2, nullptr, (float*)d_out, N2v, OUTv, OUTv, OUTv);
}
